// QuadraticConv2D_73942156968363
// MI455X (gfx1250) — matrix operand order repaired and checked by reading, not compiled, not run on hardware
//
#include <hip/hip_runtime.h>

typedef _Float16 v16h __attribute__((ext_vector_type(16)));
typedef _Float16 v8h  __attribute__((ext_vector_type(8)));
typedef float    v8f  __attribute__((ext_vector_type(8)));

#define Bsz   16
#define Hsz   64
#define Wsz   64
#define Cin   64
#define Cout  64
#define Lfeat 55
#define HP    66   // padded H
#define WP    66   // padded W

// d_ws layout
#define XPAD_OFF    0ul
#define WREP_OFF    (9ul * 1024 * 1024)                 // padded-x is 8,921,088 B
#define CHALF_BYTES ((size_t)Lfeat * 4 * 32 * 16 * 2)   // 225,280 B per channel-half
#define LGRP        11                                  // features per chunk
#define NFRAG       (LGRP * 4)                          // 44 B-fragments per chunk
#define CHUNK_BYTES (LGRP * 4 * 1024)                   // 45,056 B
#define LDS_BYTES   (3 * CHUNK_BYTES)                   // 135,168 B (triple buffer)

// triu pair tables (i<=j over 9 taps), compile-time folded after full unroll
__device__ constexpr int IU[45] = {0,0,0,0,0,0,0,0,0, 1,1,1,1,1,1,1,1, 2,2,2,2,2,2,2,
                                   3,3,3,3,3,3, 4,4,4,4,4, 5,5,5,5, 6,6,6, 7,7, 8};
__device__ constexpr int JU[45] = {0,1,2,3,4,5,6,7,8, 1,2,3,4,5,6,7,8, 2,3,4,5,6,7,8,
                                   3,4,5,6,7,8, 4,5,6,7,8, 5,6,7,8, 6,7,8, 7,8, 8};

// ---------------- pre-pass 1: pad + convert x (f32 NHWC) -> f16 [B][66][66][64] --------
__global__ void pad_cvt_kernel(const float* __restrict__ x, _Float16* __restrict__ xpad) {
    int idx = blockIdx.x * 256 + threadIdx.x;           // one thread = 8 channels of one padded pixel
    int cg = idx & 7;
    int p  = idx >> 3;
    int wp = p % WP;  p /= WP;
    int hp = p % HP;
    int b  = p / HP;
    v8h val = {};
    if (hp != 0 && hp != HP - 1 && wp != 0 && wp != WP - 1) {
        const float* src = x + ((size_t)((b * Hsz + (hp - 1)) * Wsz + (wp - 1))) * Cin + cg * 8;
        #pragma unroll
        for (int i = 0; i < 8; ++i) val[i] = (_Float16)src[i];
    }
    *(v8h*)(xpad + ((size_t)(b * HP + hp) * WP + wp) * Cin + cg * 8) = val;
}

// ---------------- pre-pass 2: repack weights [L][Cin][Cout] f32 -> f16 WMMA-B fragments -
// Fragment (chalf, l, nt) is a 32x16 f16 B tile: B[kk][n] = w[l][chalf*32+kk][nt*16+n].
// ISA B layout (16-bit, 32x16): lane holds N=lane&15; halfs 0..7 -> K=hi*8+0..7,
// halfs 8..15 -> K=16+hi*8+0..7 (hi=lane>>4). Each lane's 16 halfs stored contiguously.
__global__ void wrep_kernel(const float* __restrict__ w, _Float16* __restrict__ wrep) {
    int idx = blockIdx.x * 256 + threadIdx.x;           // one thread = one half
    int half = idx & 15;
    int lane = (idx >> 4) & 31;
    int f    = idx >> 9;                                 // fragment id = (chalf*Lfeat + l)*4 + nt
    int nt   = f & 3;  f >>= 2;
    int l    = f % Lfeat;
    int chalf = f / Lfeat;
    int n  = lane & 15;
    int hi = lane >> 4;
    int kk = (half < 8) ? hi * 8 + half : 16 + hi * 8 + (half - 8);
    int c  = chalf * 32 + kk;
    int o  = nt * 16 + n;
    wrep[idx] = (_Float16)w[((size_t)l * Cin + c) * Cout + o];
}

// ---------------- async global -> LDS staging (ASYNCcnt-tracked DMA path) --------------
__device__ __forceinline__ void async_ld16(unsigned lds_off, unsigned voff, const void* saddr) {
    asm volatile("global_load_async_to_lds_b128 %0, %1, %2"
                 :: "v"(lds_off), "v"(voff), "s"(saddr)
                 : "memory");
}

__device__ __forceinline__ void issue_chunk(const char* wrep, int ch, int lg, int bufsel, int tid) {
    const char* src = wrep + (size_t)ch * CHALF_BYTES + (size_t)lg * CHUNK_BYTES;
    unsigned lds0 = (unsigned)(bufsel * CHUNK_BYTES) + (unsigned)tid * 16u;
    unsigned voff = (unsigned)tid * 16u;
    #pragma unroll
    for (int i = 0; i < LGRP; ++i)                      // 256 thr x 16 B x 11 = 45,056 B
        async_ld16(lds0 + i * 4096u, voff + i * 4096u, src);
}

__device__ __forceinline__ v16h ldB(const char* buf, int fid, int lane) {
    return *(const v16h*)(buf + fid * 1024 + lane * 32);
}

__global__ __launch_bounds__(256, 2)
void qconv_main(const _Float16* __restrict__ xpad,
                const _Float16* __restrict__ wrep,
                float* __restrict__ out) {
    extern __shared__ char smem[];

    const int tid  = threadIdx.x;
    const int lane = tid & 31;
    const int wave = tid >> 5;
    const int pair = blockIdx.x * 8 + wave;   // 2048 tile-pairs total
    const int wh   = pair & 1;                // which half-row (w 0..31 or 32..63)
    const int row  = pair >> 1;
    const int h    = row & (Hsz - 1);
    const int b    = row >> 6;
    const int w0   = wh * 32;
    const int m    = lane & 15;
    const int hi   = lane >> 4;

    v8f acc0[4] = {};
    v8f acc1[4] = {};

    // prologue: kick off chunk 0 (chalf 0, lgroup 0) into LDS buffer 0
    issue_chunk((const char*)wrep, 0, 0, 0, tid);

    for (int chalf = 0; chalf < 2; ++chalf) {
        // 2 x 9 patch A-fragments (global loads overlap outstanding async DMA).
        // ISA 16-bit A layout: halfs 0..7 -> channels c0+0..7 (K = hi*8 + h),
        //                      halfs 8..15 -> channels c0+16+0..7 (K = 16 + hi*8 + h)
        const int c0 = chalf * 32 + hi * 8;
        v16h pat0[9], pat1[9];
        #pragma unroll
        for (int di = 0; di < 3; ++di)
            #pragma unroll
            for (int dj = 0; dj < 3; ++dj) {
                const _Float16* px0 =
                    xpad + ((size_t)(b * HP + h + di) * WP + (w0 + m + dj)) * Cin;
                const _Float16* px1 = px0 + (size_t)16 * Cin;   // second tile: w + 16
                v8h lo0  = *(const v8h*)(px0 + c0);
                v8h hi0  = *(const v8h*)(px0 + c0 + 16);
                v8h lo1  = *(const v8h*)(px1 + c0);
                v8h hi1  = *(const v8h*)(px1 + c0 + 16);
                v16h f0, f1;
                #pragma unroll
                for (int i = 0; i < 8; ++i) {
                    f0[i] = lo0[i]; f0[i + 8] = hi0[i];
                    f1[i] = lo1[i]; f1[i + 8] = hi1[i];
                }
                pat0[di * 3 + dj] = f0;
                pat1[di * 3 + dj] = f1;
            }

        #pragma unroll
        for (int cc = 0; cc < 5; ++cc) {                 // 5 chunks of 11 features
            const int g = chalf * 5 + cc;                // global chunk id 0..9

            // chunk g's async loads (issued one chunk ago) must be LDS-resident;
            // single barrier per chunk: triple buffering makes the write target
            // (buffer (g+1)%3, last read at chunk g-2) provably free after this barrier.
            asm volatile("s_wait_asynccnt 0x0" ::: "memory");
            __syncthreads();

            // prefetch next chunk (runs under the WMMAs below)
            if (g < 9) {
                const int gn = g + 1;
                issue_chunk((const char*)wrep, gn >= 5 ? 1 : 0, gn >= 5 ? gn - 5 : gn,
                            gn % 3, tid);
            }

            const char* buf = smem + (g % 3) * CHUNK_BYTES;

            // explicit 3-deep rotating register pipeline for B fragments:
            // consume fid, prefetch fid+2 (4 ds_load_b128 always in flight)
            v16h bq0 = ldB(buf, 0, lane);
            v16h bq1 = ldB(buf, 1, lane);
            v16h bq2;
            #pragma unroll
            for (int ll = 0; ll < LGRP; ++ll) {
                const int l = cc * LGRP + ll;            // compile-time after unroll
                v16h a0, a1;
                if (l < 45)      { a0 = pat0[IU[l]] * pat0[JU[l]];
                                   a1 = pat1[IU[l]] * pat1[JU[l]]; }
                else if (l < 54) { a0 = pat0[l - 45]; a1 = pat1[l - 45]; }
                else             { v16h o1;
                                   #pragma unroll
                                   for (int i = 0; i < 16; ++i) o1[i] = (_Float16)1.0f;
                                   a0 = o1; a1 = o1; }
                #pragma unroll
                for (int nt = 0; nt < 4; ++nt) {
                    const int fid = ll * 4 + nt;
                    const v16h bcur = (fid % 3 == 0) ? bq0 : (fid % 3 == 1) ? bq1 : bq2;
                    if (fid + 2 < NFRAG) {
                        const v16h bn = ldB(buf, fid + 2, lane);
                        if      ((fid + 2) % 3 == 0) bq0 = bn;
                        else if ((fid + 2) % 3 == 1) bq1 = bn;
                        else                         bq2 = bn;
                    }
                    acc0[nt] = __builtin_amdgcn_wmma_f32_16x16x32_f16(
                        false, a0, false, bcur, (short)0, acc0[nt], false, false);
                    acc1[nt] = __builtin_amdgcn_wmma_f32_16x16x32_f16(
                        false, a1, false, bcur, (short)0, acc1[nt], false, false);
                }
            }
        }
    }

    // C/D layout: VGPR r -> M = hi*8 + r, N = lane&15
    #pragma unroll
    for (int nt = 0; nt < 4; ++nt)
        #pragma unroll
        for (int r = 0; r < 8; ++r) {
            int M = hi * 8 + r;
            size_t base = ((size_t)(b * Hsz + h) * Wsz);
            out[(base + (w0 + M))      * Cout + nt * 16 + (lane & 15)] = acc0[nt][r];
            out[(base + (w0 + 16 + M)) * Cout + nt * 16 + (lane & 15)] = acc1[nt][r];
        }
}

extern "C" void kernel_launch(void* const* d_in, const int* in_sizes, int n_in,
                              void* d_out, int out_size, void* d_ws, size_t ws_size,
                              hipStream_t stream) {
    const float* x  = (const float*)d_in[0];
    const float* fk = (const float*)d_in[1];   // [1,1,55,64,64] f32
    float* out = (float*)d_out;
    char*  ws  = (char*)d_ws;

    _Float16* xpad = (_Float16*)(ws + XPAD_OFF);
    _Float16* wrep = (_Float16*)(ws + WREP_OFF);

    // 16*66*66*8 = 557,568 threads -> 2178 blocks
    pad_cvt_kernel<<<2178, 256, 0, stream>>>(x, xpad);
    // 450,560 threads -> 1760 blocks
    wrep_kernel<<<1760, 256, 0, stream>>>(fk, wrep);
    // 2048 tile-pairs / 8 waves per block -> 256 blocks, 132 KB dynamic LDS (triple buffer)
    qconv_main<<<256, 256, LDS_BYTES, stream>>>(xpad, wrep, out);
}
